// EdgeWeight_6090263625943
// MI455X (gfx1250) — matrix operand order repaired and checked by reading, not compiled, not run on hardware
//
#include <hip/hip_runtime.h>
#include <hip/hip_bf16.h>

typedef unsigned short ushort_t;
typedef unsigned int uint_t;

typedef __attribute__((ext_vector_type(16))) __bf16 v16bf;
typedef __attribute__((ext_vector_type(8)))  float  v8f;
typedef __attribute__((ext_vector_type(4)))  float  v4f;   // clang ext vector: legal for nontemporal builtins

union FragU { v16bf v; uint4 q[2]; };

#define DIM 512
#define NROW 36
#define NB 64
#define M_TOT (NB * NROW * NROW)   // 82944 rows
#define K_REDUCE 12

__device__ __forceinline__ ushort_t f2bf(float f) {
    uint_t u = __float_as_uint(f);
    // round-to-nearest-even bf16 truncation
    u = (u + 0x7FFFu + ((u >> 16) & 1u)) >> 16;
    return (ushort_t)u;
}

// ---------------------------------------------------------------------------
// Prep: transpose + convert fp32 weight [K][N] -> bf16 [N][K] so WMMA B-matrix
// fragments are 32-byte contiguous per lane (column n, K contiguous).
// ---------------------------------------------------------------------------
__global__ void transpose_bf16_kernel(const float* __restrict__ in,
                                      ushort_t* __restrict__ out,
                                      int K, int N) {
    int idx = blockIdx.x * blockDim.x + threadIdx.x;
    if (idx >= K * N) return;
    int n = idx / K;
    int k = idx - n * K;
    out[n * K + k] = f2bf(in[k * N + n]);
}

// ---------------------------------------------------------------------------
// Fused edge-MLP: 16 rows per workgroup, 256 threads (8 wave32).
//   GEMM1 (bf16 WMMA, 16x512 * 512x512) + b1
//   LayerNorm over D (shuffle + LDS reduction)
//   GEMM2 (bf16 WMMA, 16x512 * 512x256) + b2
//   GLU (a * sigmoid(g)) via LDS exchange
//   GEMM3 (dot with W3[128]) -> scores
// e_feats is a 170MB single-use stream -> non-temporal loads so the
// all-blocks-shared bf16 weights stay resident in the 192MB L2.
// ---------------------------------------------------------------------------
__global__ __launch_bounds__(256) void edge_mlp_kernel(
    const float* __restrict__ e_feats,
    const ushort_t* __restrict__ W1t,   // bf16 [512][512] (n-major)
    const float* __restrict__ b1,
    const float* __restrict__ ln_g,
    const float* __restrict__ ln_b,
    const ushort_t* __restrict__ W2t,   // bf16 [256][512] (n-major)
    const float* __restrict__ b2,
    const float* __restrict__ W3,       // f32 [128]
    const float* __restrict__ b3,       // f32 [1]
    float* __restrict__ scores)
{
    __shared__ ushort_t sA[16 * DIM];   // 16KB: bf16 A tile; later reused as f32 sOut[16][256]
    __shared__ ushort_t sH[16 * DIM];   // 16KB: normalized h (bf16), A-matrix for GEMM2
    __shared__ float sRedS[8 * 16];
    __shared__ float sRedQ[8 * 16];
    __shared__ float sMu[16];
    __shared__ float sRs[16];

    const int tid  = threadIdx.x;
    const int wave = tid >> 5;
    const int lane = tid & 31;
    const int g    = lane >> 4;      // lane half (WMMA layout group)
    const int lr   = lane & 15;
    const int kA   = g * 16;         // A-frag K offset per lane half (contiguous K, matches B)
    const int kB   = g * 16;         // B-frag K offset per lane half (contiguous K, matches A)

    const long long rowBase = (long long)blockIdx.x * 16;
    const float* src = e_feats + rowBase * DIM;

    // ---- stage A tile: 16x512 f32 -> bf16 in LDS (NT streamed float4 loads) ----
    #pragma unroll
    for (int i = 0; i < 8; ++i) {
        int linear = i * 1024 + tid * 4;
        v4f f = __builtin_nontemporal_load((const v4f*)(src + linear));
        uint2 p;
        p.x = (uint_t)f2bf(f.x) | ((uint_t)f2bf(f.y) << 16);
        p.y = (uint_t)f2bf(f.z) | ((uint_t)f2bf(f.w) << 16);
        *(uint2*)(&sA[linear]) = p;
    }
    __syncthreads();

    // ---- GEMM1: each wave owns 4 column tiles (64 cols), K = 512 in 16 steps ----
    v8f acc[4] = {};
    for (int k0 = 0; k0 < DIM; k0 += 32) {
        FragU a;
        a.q[0] = *(const uint4*)(&sA[lr * DIM + k0 + kA]);
        a.q[1] = *(const uint4*)(&sA[lr * DIM + k0 + kA + 8]);
        #pragma unroll
        for (int t = 0; t < 4; ++t) {
            int col = wave * 64 + t * 16 + lr;
            FragU b;
            const uint4* bp = (const uint4*)(&W1t[col * DIM + k0 + kB]);
            b.q[0] = bp[0];
            b.q[1] = bp[1];
            acc[t] = __builtin_amdgcn_wmma_f32_16x16x32_bf16(
                false, a.v, false, b.v, (short)0, acc[t], false, false);
        }
    }

    // ---- + b1 ----
    #pragma unroll
    for (int t = 0; t < 4; ++t) {
        float bv = b1[wave * 64 + t * 16 + lr];
        #pragma unroll
        for (int j = 0; j < 8; ++j) acc[t][j] += bv;
    }

    // ---- LayerNorm statistics: per-row sum / sumsq over 512 columns ----
    // C-layout: lanes 0-15 hold rows j (vgpr j), lanes 16-31 hold rows j+8.
    #pragma unroll
    for (int j = 0; j < 8; ++j) {
        float s = 0.f, q = 0.f;
        #pragma unroll
        for (int t = 0; t < 4; ++t) { float x = acc[t][j]; s += x; q += x * x; }
        #pragma unroll
        for (int m = 1; m < 16; m <<= 1) {
            s += __shfl_xor(s, m, 16);
            q += __shfl_xor(q, m, 16);
        }
        if (lr == 0) {
            sRedS[wave * 16 + j + 8 * g] = s;
            sRedQ[wave * 16 + j + 8 * g] = q;
        }
    }
    __syncthreads();
    if (tid < 16) {
        float s = 0.f, q = 0.f;
        #pragma unroll
        for (int w = 0; w < 8; ++w) { s += sRedS[w * 16 + tid]; q += sRedQ[w * 16 + tid]; }
        float mu  = s * (1.0f / 512.0f);
        float var = q * (1.0f / 512.0f) - mu * mu;
        sMu[tid] = mu;
        sRs[tid] = rsqrtf(var + 1e-5f);
    }
    __syncthreads();

    // ---- normalize, scale/shift, write bf16 h to LDS (row-major, A-matrix) ----
    #pragma unroll
    for (int t = 0; t < 4; ++t) {
        int col = wave * 64 + t * 16 + lr;
        float gg = ln_g[col];
        float bb = ln_b[col];
        #pragma unroll
        for (int j = 0; j < 8; ++j) {
            int m = j + 8 * g;
            float x = (acc[t][j] - sMu[m]) * sRs[m] * gg + bb;
            sH[m * DIM + col] = f2bf(x);
        }
    }
    __syncthreads();

    // ---- GEMM2: each wave owns 2 column tiles (32 of 256 cols), K = 512 ----
    v8f acc2[2] = {};
    for (int k0 = 0; k0 < DIM; k0 += 32) {
        FragU a;
        a.q[0] = *(const uint4*)(&sH[lr * DIM + k0 + kA]);
        a.q[1] = *(const uint4*)(&sH[lr * DIM + k0 + kA + 8]);
        #pragma unroll
        for (int t = 0; t < 2; ++t) {
            int col = wave * 32 + t * 16 + lr;
            FragU b;
            const uint4* bp = (const uint4*)(&W2t[col * DIM + k0 + kB]);
            b.q[0] = bp[0];
            b.q[1] = bp[1];
            acc2[t] = __builtin_amdgcn_wmma_f32_16x16x32_bf16(
                false, a.v, false, b.v, (short)0, acc2[t], false, false);
        }
    }

    // ---- + b2, spill 16x256 f32 to LDS (reuse sA region) for GLU exchange ----
    float* sOut = (float*)sA;   // 16*256 f32 = 16KB, sA no longer read
    #pragma unroll
    for (int t = 0; t < 2; ++t) {
        int col = wave * 32 + t * 16 + lr;
        float bv = b2[col];
        #pragma unroll
        for (int j = 0; j < 8; ++j) {
            int m = j + 8 * g;
            sOut[m * 256 + col] = acc2[t][j] + bv;
        }
    }
    __syncthreads();

    // ---- GLU + GEMM3: scores[m] = sum_c a[m][c]*sigmoid(g[m][c]) * W3[c] + b3 ----
    {
        int m  = tid >> 4;
        int c0 = (tid & 15) * 8;
        float p = 0.f;
        #pragma unroll
        for (int c = c0; c < c0 + 8; ++c) {
            float av = sOut[m * 256 + c];
            float gv = sOut[m * 256 + 128 + c];
            float glu = av * (1.0f / (1.0f + __expf(-gv)));
            p += glu * W3[c];
        }
        p += __shfl_down(p, 8, 16);
        p += __shfl_down(p, 4, 16);
        p += __shfl_down(p, 2, 16);
        p += __shfl_down(p, 1, 16);
        if ((tid & 15) == 0) scores[rowBase + m] = p + b3[0];
    }
}

// ---------------------------------------------------------------------------
// Softmax over the neighbor axis (36) + top-12 keep (threshold = 12th largest
// softmax weight) + always-keep diagonal. One thread per (b, i) row: tiny.
// ---------------------------------------------------------------------------
__global__ void softmax_topk_kernel(const float* __restrict__ scores,
                                    float* __restrict__ out) {
    int row = blockIdx.x * blockDim.x + threadIdx.x;
    if (row >= NB * NROW) return;
    const float* s = scores + row * NROW;
    float v[NROW];
    float mx = -3.0e38f;
    for (int j = 0; j < NROW; ++j) { v[j] = s[j]; mx = fmaxf(mx, v[j]); }
    float sum = 0.f;
    for (int j = 0; j < NROW; ++j) { v[j] = __expf(v[j] - mx); sum += v[j]; }
    float inv = 1.0f / sum;
    for (int j = 0; j < NROW; ++j) v[j] *= inv;
    // 12th-largest threshold via iterative selection
    float t[NROW];
    for (int j = 0; j < NROW; ++j) t[j] = v[j];
    float thresh = 0.f;
    for (int it = 0; it < K_REDUCE; ++it) {
        float m = -1.f; int mi = 0;
        for (int j = 0; j < NROW; ++j) if (t[j] > m) { m = t[j]; mi = j; }
        thresh = m;
        t[mi] = -2.f;
    }
    int diag = row % NROW;   // self edge j == i
    float* o = out + row * NROW;
    for (int j = 0; j < NROW; ++j) {
        float val = (v[j] >= thresh || j == diag) ? v[j] : 0.f;
        __builtin_nontemporal_store(val, o + j);   // write-once output
    }
}

// ---------------------------------------------------------------------------
extern "C" void kernel_launch(void* const* d_in, const int* in_sizes, int n_in,
                              void* d_out, int out_size, void* d_ws, size_t ws_size,
                              hipStream_t stream) {
    const float* e_feats = (const float*)d_in[0];
    const float* W1      = (const float*)d_in[1];
    const float* b1      = (const float*)d_in[2];
    const float* ln_g    = (const float*)d_in[3];
    const float* ln_b    = (const float*)d_in[4];
    const float* W2      = (const float*)d_in[5];
    const float* b2      = (const float*)d_in[6];
    const float* W3      = (const float*)d_in[7];
    const float* b3      = (const float*)d_in[8];
    float* out = (float*)d_out;

    char* ws = (char*)d_ws;
    ushort_t* W1t   = (ushort_t*)ws;                                   // 512*512*2 = 512KB
    ushort_t* W2t   = (ushort_t*)(ws + DIM * DIM * 2);                 // 256*512*2 = 256KB
    float*    score = (float*)(ws + DIM * DIM * 2 + 256 * DIM * 2);    // 82944*4

    // weight prep (cheap, L2-resident afterwards)
    transpose_bf16_kernel<<<(DIM * DIM + 255) / 256, 256, 0, stream>>>(W1, W1t, DIM, DIM);
    transpose_bf16_kernel<<<(DIM * 256 + 255) / 256, 256, 0, stream>>>(W2, W2t, DIM, 256);

    // fused MLP over all 82944 edge rows, 16 rows per block
    edge_mlp_kernel<<<M_TOT / 16, 256, 0, stream>>>(
        e_feats, W1t, b1, ln_g, ln_b, W2t, b2, W3, b3, score);

    // softmax + top-k mask
    softmax_topk_kernel<<<(NB * NROW + 255) / 256, 256, 0, stream>>>(score, out);
}
